// DynamicKnowledgeInjector_71270687309849
// MI455X (gfx1250) — compile-verified, matrix operand order checked by reading, not run on hardware
//
#include <hip/hip_runtime.h>
#include <hip/hip_bf16.h>
#include <stdint.h>

// ---------------------------------------------------------------------------
// DynamicKnowledgeInjector for MI455X (gfx1250), wave32 + WMMA bf16 pipeline.
// B=8 T=2048 H=1024 E=768 F=64 R=C(64,2)=2016 TOP_K=28, BT = 16384.
// ---------------------------------------------------------------------------

typedef __attribute__((ext_vector_type(16))) __bf16 v16bf;
typedef __attribute__((ext_vector_type(8)))  float  v8f;

#define WAVE   32
#define HDIM   1024
#define EDIM   768
#define RNUM   2016
#define NTILE  126      // RNUM / 16
#define BT     16384
#define TOPK   28

union Frag16 { v16bf v; uint4 u[2]; };

// Load 16 bf16 (one WMMA A/B fragment for this lane) from two 16B-aligned
// 8-element groups: k = {base, base+16} + hi*8 .. +7  (ISA 16-bit 16x32 layout)
__device__ __forceinline__ void loadFrag(Frag16& f, const __bf16* p0, const __bf16* p1) {
  f.u[0] = *(const uint4*)p0;
  f.u[1] = *(const uint4*)p1;
}
__device__ __forceinline__ void loadFrag(Frag16& f, const float* p0, const float* p1) {
  __bf16* d = (__bf16*)&f;
  float4 a0 = ((const float4*)p0)[0], a1 = ((const float4*)p0)[1];
  float4 b0 = ((const float4*)p1)[0], b1 = ((const float4*)p1)[1];
  d[0]=(__bf16)a0.x; d[1]=(__bf16)a0.y; d[2]=(__bf16)a0.z; d[3]=(__bf16)a0.w;
  d[4]=(__bf16)a1.x; d[5]=(__bf16)a1.y; d[6]=(__bf16)a1.z; d[7]=(__bf16)a1.w;
  d[8]=(__bf16)b0.x; d[9]=(__bf16)b0.y; d[10]=(__bf16)b0.z; d[11]=(__bf16)b0.w;
  d[12]=(__bf16)b1.x; d[13]=(__bf16)b1.y; d[14]=(__bf16)b1.z; d[15]=(__bf16)b1.w;
}

__device__ __forceinline__ float waveMax(float v) {
  #pragma unroll
  for (int o = 16; o > 0; o >>= 1) v = fmaxf(v, __shfl_xor(v, o, 32));
  return v;
}
__device__ __forceinline__ float waveSum(float v) {
  #pragma unroll
  for (int o = 16; o > 0; o >>= 1) v += __shfl_xor(v, o, 32);
  return v;
}

// ---------------------------------------------------------------------------
// f32 -> bf16 elementwise convert (for the small weight matrices)
// ---------------------------------------------------------------------------
__global__ void cvt_bf16(const float* __restrict__ in, __bf16* __restrict__ outp, int n) {
  int i = blockIdx.x * blockDim.x + threadIdx.x;
  if (i < n) outp[i] = (__bf16)in[i];
}

// ---------------------------------------------------------------------------
// Generic WMMA GEMM: C[M,N] = A[M,Kd] * W[N,Kd]^T + bias[N]
// A is f32 or bf16 (converted in-registers), W is bf16, C is bf16 or f32.
// One wave computes a 16x64 output tile (A fragment reused across 4 WMMAs).
// grid: (M/16)*(N/64) waves, 8 waves / block.
// ---------------------------------------------------------------------------
template<typename AT, typename OT>
__global__ __launch_bounds__(256) void gemm_bias(
    const AT* __restrict__ A, const __bf16* __restrict__ W,
    const float* __restrict__ bias, OT* __restrict__ C,
    int M, int N, int Kd) {
  const int wv   = (blockIdx.x * blockDim.x + threadIdx.x) / WAVE;
  const int lane = threadIdx.x & (WAVE - 1);
  const int lo   = lane & 15, hi = lane >> 4;
  const int ntg  = N >> 6;                 // column groups of 64
  const int mt   = wv / ntg, ng = wv % ntg;
  if (mt * 16 >= M) return;

  const AT*     arow = A + (size_t)(mt * 16 + lo) * Kd + hi * 8;
  const __bf16* brow = W + (size_t)(ng * 64 + lo) * Kd + hi * 8;

  v8f acc[4] = {};
  for (int kk = 0; kk < Kd; kk += 32) {
    Frag16 fa;
    loadFrag(fa, arow + kk, arow + kk + 16);
    #pragma unroll
    for (int j = 0; j < 4; ++j) {
      Frag16 fb;
      const __bf16* bj = brow + (size_t)(j * 16) * Kd;
      loadFrag(fb, bj + kk, bj + kk + 16);
      acc[j] = __builtin_amdgcn_wmma_f32_16x16x32_bf16(
          false, fa.v, false, fb.v, (short)0, acc[j], false, false);
    }
  }
  // D layout: lane = n%16 + 16*(m/8); vgpr r -> row r + 8*(lane/16)
  #pragma unroll
  for (int j = 0; j < 4; ++j) {
    const int nn = ng * 64 + j * 16 + lo;
    const float bb = bias[nn];
    #pragma unroll
    for (int r = 0; r < 8; ++r) {
      const int mm = mt * 16 + r + hi * 8;
      C[(size_t)mm * N + nn] = (OT)(acc[j][r] + bb);
    }
  }
}

// ---------------------------------------------------------------------------
// Attention kernel: per 16-row tile of BT, compute 16x2016 scores via WMMA in
// LDS, mask by relation activity, exact top-28 threshold per row, softmax,
// sparse attn@V, write f32 output. 256 threads = 8 waves per workgroup.
// ---------------------------------------------------------------------------
__global__ __launch_bounds__(256) void attn_topk(
    const __bf16* __restrict__ Qb, const __bf16* __restrict__ Kb,
    const float* __restrict__ Vf, const unsigned char* __restrict__ surv,
    const int* __restrict__ fi, const int* __restrict__ fj,
    float* __restrict__ out) {
  __shared__ float sS[16][RNUM];                       // 129 KB score tile
  __shared__ union {
    __bf16 q[16][HDIM];                                // 32 KB Q staging
    struct { int r[16][TOPK]; float p[16][TOPK]; } sel; // recycled after phase 1
  } U;
  __shared__ unsigned long long rmask[16];
  __shared__ float sMax[16], sDen[16], sThr[16];
  __shared__ int   sCnt[16];

  const int tid  = threadIdx.x;
  const int w    = tid >> 5, lane = tid & 31;
  const int lo   = lane & 15, hi = lane >> 4;
  const int row0 = blockIdx.x * 16;
  const float NEGINF = -__builtin_inff();

  // ---- phase 0: stage Q tile (bf16) + build per-row 64-bit feature masks ----
  {
    const uint4* src = (const uint4*)(Qb + (size_t)row0 * HDIM);
    uint4* dst = (uint4*)&U.q[0][0];
    for (int i = tid; i < 16 * HDIM * 2 / 16; i += 256) dst[i] = src[i];
  }
  if (tid < 16) {
    const unsigned char* mrow = surv + (size_t)(row0 + tid) * 64;
    unsigned long long mk = 0ULL;
    #pragma unroll 8
    for (int f = 0; f < 64; ++f) mk |= (unsigned long long)(mrow[f] != 0) << f;
    rmask[tid] = mk;
  }
  __syncthreads();

  // ---- phase 1: scores = (Q tile) x K^T via WMMA, masked, into LDS ----------
  const float kinv = 0.03125f;   // 1/sqrt(H)
  for (int nt = w; nt < NTILE; nt += 8) {
    const int r = nt * 16 + lo;
    const int fia = fi[r], fja = fj[r];
    const __bf16* krow = Kb + (size_t)r * HDIM + hi * 8;
    { // prefetch next K tile for this wave (gfx1250 global_prefetch_b8)
      const int nn = nt + 8;
      if (nn < NTILE) __builtin_prefetch(Kb + (size_t)(nn * 16 + lo) * HDIM + hi * 8, 0, 3);
    }
    v8f acc = {};
    #pragma unroll 4
    for (int kk = 0; kk < HDIM; kk += 32) {
      Frag16 fa, fb;
      fa.u[0] = *(const uint4*)&U.q[lo][kk + hi * 8];
      fa.u[1] = *(const uint4*)&U.q[lo][kk + 16 + hi * 8];
      fb.u[0] = *(const uint4*)(krow + kk);
      fb.u[1] = *(const uint4*)(krow + kk + 16);
      acc = __builtin_amdgcn_wmma_f32_16x16x32_bf16(
          false, fa.v, false, fb.v, (short)0, acc, false, false);
    }
    #pragma unroll
    for (int r8 = 0; r8 < 8; ++r8) {
      const int m = r8 + hi * 8;
      const unsigned long long mk = rmask[m];
      const bool act = ((mk >> fia) & 1ULL) && ((mk >> fja) & 1ULL);
      sS[m][nt * 16 + lo] = act ? acc[r8] * kinv : NEGINF;
    }
  }
  __syncthreads();

  // ---- phase 2: per-row exact 28th-largest threshold + softmax stats --------
  // wave w owns rows 2w and 2w+1 (8 waves x 2 = 16 rows)
  for (int rr = 0; rr < 2; ++rr) {
    const int m = 2 * w + rr;
    float bound = __builtin_inff();
    float rmax  = NEGINF;
    for (int it = 0; it < TOPK; ++it) {
      float loc = NEGINF;
      for (int c = lane; c < RNUM; c += 32) {
        const float v = sS[m][c];
        if (v < bound) loc = fmaxf(loc, v);
      }
      loc = waveMax(loc);
      if (loc == NEGINF) break;      // fewer than TOPK active relations
      if (it == 0) rmax = loc;
      bound = loc;                   // after it+1 iters: (it+1)-th largest
    }
    float den = 0.f;
    if (rmax > NEGINF) {
      float s = 0.f;
      for (int c = lane; c < RNUM; c += 32) {
        const float v = sS[m][c];
        if (v >= bound && v > NEGINF) s += __expf(v - rmax);
      }
      den = waveSum(s);
    }
    if (lane == 0) {
      sThr[m] = bound; sMax[m] = rmax;
      sDen[m] = den > 0.f ? den : 1.f;
      sCnt[m] = 0;
    }
    // compact selected (index, prob) pairs into LDS (recycled Q region)
    if (rmax > NEGINF) {
      const float deni = 1.0f / (den > 0.f ? den : 1.f);
      for (int c = lane; c < RNUM; c += 32) {
        const float v = sS[m][c];
        if (v >= bound && v > NEGINF) {
          const int slot = atomicAdd(&sCnt[m], 1);
          if (slot < TOPK) {
            U.sel.r[m][slot] = c;
            U.sel.p[m][slot] = __expf(v - rmax) * deni;
          }
        }
      }
    }
  }
  __syncthreads();

  // ---- phase 3: sparse out = attn @ V  (28 V-rows per output row) -----------
  const int h0 = tid * 4;                       // 256 threads x float4 = 1024
  float* obase = out + (size_t)row0 * HDIM;
  for (int m = 0; m < 16; ++m) {
    float4 a = make_float4(0.f, 0.f, 0.f, 0.f);
    if (sMax[m] > NEGINF) {
      int n = sCnt[m]; if (n > TOPK) n = TOPK;
      for (int s = 0; s < n; ++s) {
        const int   r = U.sel.r[m][s];
        const float p = U.sel.p[m][s];
        const float4 vv = *(const float4*)(Vf + (size_t)r * HDIM + h0);
        a.x += p * vv.x; a.y += p * vv.y; a.z += p * vv.z; a.w += p * vv.w;
      }
    }
    *(float4*)(obase + (size_t)m * HDIM + h0) = a;
  }
}

// ---------------------------------------------------------------------------
// Host launcher
// ---------------------------------------------------------------------------
extern "C" void kernel_launch(void* const* d_in, const int* in_sizes, int n_in,
                              void* d_out, int out_size, void* d_ws, size_t ws_size,
                              hipStream_t stream) {
  (void)in_sizes; (void)n_in; (void)out_size; (void)ws_size;
  const float*         qh   = (const float*)d_in[0];          // [BT, H]
  const unsigned char* surv = (const unsigned char*)d_in[1];  // [BT, 64] bool
  const float*         rel  = (const float*)d_in[2];          // [R, E]
  const int*           fi   = (const int*)d_in[3];            // [R]
  const int*           fj   = (const int*)d_in[4];            // [R]
  const float* Wt = (const float*)d_in[5];  const float* bt = (const float*)d_in[6];
  const float* Wq = (const float*)d_in[7];  const float* bq = (const float*)d_in[8];
  const float* Wk = (const float*)d_in[9];  const float* bk = (const float*)d_in[10];
  const float* Wv = (const float*)d_in[11]; const float* bv = (const float*)d_in[12];
  float* out = (float*)d_out;

  char* ws = (char*)d_ws;
  size_t off = 0;
  auto alloc = [&](size_t bytes) -> void* {
    void* p = ws + off; off += (bytes + 255) & ~(size_t)255; return p;
  };
  __bf16* Wt_b = (__bf16*)alloc((size_t)HDIM * EDIM * 2);
  __bf16* Wq_b = (__bf16*)alloc((size_t)HDIM * HDIM * 2);
  __bf16* Wk_b = (__bf16*)alloc((size_t)HDIM * HDIM * 2);
  __bf16* Wv_b = (__bf16*)alloc((size_t)HDIM * HDIM * 2);
  __bf16* Ta_b = (__bf16*)alloc((size_t)RNUM * HDIM * 2);
  __bf16* K_b  = (__bf16*)alloc((size_t)RNUM * HDIM * 2);
  float*  V_f  = (float*) alloc((size_t)RNUM * HDIM * 4);
  __bf16* Q_b  = (__bf16*)alloc((size_t)BT   * HDIM * 2);

  // weight converts (small, bandwidth-trivial)
  cvt_bf16<<<(HDIM * EDIM + 255) / 256, 256, 0, stream>>>(Wt, Wt_b, HDIM * EDIM);
  cvt_bf16<<<(HDIM * HDIM + 255) / 256, 256, 0, stream>>>(Wq, Wq_b, HDIM * HDIM);
  cvt_bf16<<<(HDIM * HDIM + 255) / 256, 256, 0, stream>>>(Wk, Wk_b, HDIM * HDIM);
  cvt_bf16<<<(HDIM * HDIM + 255) / 256, 256, 0, stream>>>(Wv, Wv_b, HDIM * HDIM);

  // text_adapted = rel_embs @ Wt^T + bt     [2016,1024] bf16
  gemm_bias<float,  __bf16><<<(RNUM/16)*(HDIM/64)/8, 256, 0, stream>>>(
      rel, Wt_b, bt, Ta_b, RNUM, HDIM, EDIM);
  // K = Ta @ Wk^T + bk                      [2016,1024] bf16
  gemm_bias<__bf16, __bf16><<<(RNUM/16)*(HDIM/64)/8, 256, 0, stream>>>(
      Ta_b, Wk_b, bk, K_b, RNUM, HDIM, HDIM);
  // V = Ta @ Wv^T + bv                      [2016,1024] f32
  gemm_bias<__bf16, float ><<<(RNUM/16)*(HDIM/64)/8, 256, 0, stream>>>(
      Ta_b, Wv_b, bv, V_f, RNUM, HDIM, HDIM);
  // Q = query_hidden @ Wq^T + bq            [16384,1024] bf16
  gemm_bias<float,  __bf16><<<(BT/16)*(HDIM/64)/8, 256, 0, stream>>>(
      qh, Wq_b, bq, Q_b, BT, HDIM, HDIM);

  // scores + top-28 + softmax + sparse attn@V
  attn_topk<<<BT / 16, 256, 0, stream>>>(Q_b, K_b, V_f, surv, fi, fj, out);
}
